// AttentionWithPos_2353642078265
// MI455X (gfx1250) — matrix operand order repaired and checked by reading, not compiled, not run on hardware
//
#include <hip/hip_runtime.h>
#include <hip/hip_bf16.h>

typedef __bf16 bf16_t;
typedef bf16_t v16bf __attribute__((ext_vector_type(16)));
typedef float  v8f   __attribute__((ext_vector_type(8)));

#define NB   4
#define SEQ  4096
#define DIM  2048
#define NH   32
#define HDIM 64
#define MTOT (NB * SEQ)

// GEMM tiling
#define BM 128
#define BN 128
#define BK 32
#define LSTR 40                      // LDS row stride in bf16 elements (pad vs 32)
#define TILE_BYTES (BM * LSTR * 2)   // one LDS buffer: 10240 B

__device__ __forceinline__ v8f zero8() {
    v8f z = {0.f, 0.f, 0.f, 0.f, 0.f, 0.f, 0.f, 0.f};
    return z;
}

// ---- CDNA5 async global->LDS (ASYNCcnt) ------------------------------------
__device__ __forceinline__ void async_copy16(uint32_t lds_byte_off, const void* g) {
    asm volatile("global_load_async_to_lds_b128 %0, %1, off"
                 :: "v"(lds_byte_off), "v"((uint64_t)(uintptr_t)g)
                 : "memory");
}
__device__ __forceinline__ void wait_async0() {
    asm volatile("s_wait_asynccnt 0x0" ::: "memory");
}
__device__ __forceinline__ uint32_t lds_off(const void* p) {
    return (uint32_t)(uintptr_t)p;   // low 32 bits of a flat shared pointer = LDS offset
}

// ---- WMMA fragment loaders -------------------------------------------------
// A (16x32 bf16, M on lanes): lanes 0-15 hold K {0..7,16..23}, lanes 16-31 hold
// K {8..15,24..31} for row M = lane&15.
__device__ __forceinline__ v16bf frag_a_lds(const bf16_t* tile, int lane) {
    const bf16_t* p = tile + (lane & 15) * LSTR + ((lane & 16) ? 8 : 0);
    v16bf a;
    ((uint4*)&a)[0] = *(const uint4*)(p);
    ((uint4*)&a)[1] = *(const uint4*)(p + 16);
    return a;
}
// B (32x16 bf16, N on lanes): lanes 0-15 hold K {0..7,16..23}, lanes 16-31 hold
// K {8..15,24..31} for column N = lane&15 (rows of the LDS tile are N, cols are K).
__device__ __forceinline__ v16bf frag_b_lds(const bf16_t* tile, int lane) {
    const bf16_t* p = tile + (lane & 15) * LSTR + ((lane & 16) ? 8 : 0);
    v16bf b;
    ((uint4*)&b)[0] = *(const uint4*)(p);
    ((uint4*)&b)[1] = *(const uint4*)(p + 16);
    return b;
}
// Same layouts, loading straight from a row-major [rows][HDIM] bf16 matrix.
__device__ __forceinline__ v16bf frag_a_gmem(const bf16_t* m, int hbase, int kt, int lane) {
    const bf16_t* p = m + (size_t)(hbase + (lane & 15)) * HDIM + kt * 32 + ((lane & 16) ? 8 : 0);
    v16bf a;
    ((uint4*)&a)[0] = *(const uint4*)(p);
    ((uint4*)&a)[1] = *(const uint4*)(p + 16);
    return a;
}
__device__ __forceinline__ v16bf frag_b_gmem(const bf16_t* m, int hbase, int kt, int lane) {
    const bf16_t* p = m + (size_t)(hbase + (lane & 15)) * HDIM + kt * 32 + ((lane & 16) ? 8 : 0);
    v16bf b;
    ((uint4*)&b)[0] = *(const uint4*)(p);
    ((uint4*)&b)[1] = *(const uint4*)(p + 16);
    return b;
}

// ---- fp32 -> bf16 bulk conversion ------------------------------------------
__global__ __launch_bounds__(256) void cvt_f32_bf16(const float* __restrict__ in,
                                                    bf16_t* __restrict__ out, int n8) {
    const int i = blockIdx.x * 256 + threadIdx.x;
    if (i >= n8) return;
    const float4* f = (const float4*)in + (size_t)i * 2;
    const float4 a = f[0], b = f[1];
    union { bf16_t h[8]; uint4 q; } pk;
    pk.h[0] = (bf16_t)a.x; pk.h[1] = (bf16_t)a.y; pk.h[2] = (bf16_t)a.z; pk.h[3] = (bf16_t)a.w;
    pk.h[4] = (bf16_t)b.x; pk.h[5] = (bf16_t)b.y; pk.h[6] = (bf16_t)b.z; pk.h[7] = (bf16_t)b.w;
    ((uint4*)out)[i] = pk.q;
}

// ---- Tiled GEMM core: Out[m,n] = sum_k X[m,k] * W[n,k] + bias[n] -----------
// All-bf16 operands. 256 threads = 8 waves (2 along M x 4 along N). Each wave
// owns two 16-wide N-tiles placed 32 columns apart so the RoPE rotation partner
// (hd, hd+32) sits in the same lane of the paired accumulators.
// Double-buffered LDS filled by async global->LDS copies (ASYNCcnt pipeline).
template <bool OUT_BF16>
__device__ __forceinline__ void gemm_core(const bf16_t* __restrict__ Xb,
                                          const bf16_t* __restrict__ Wb,
                                          const float* __restrict__ bias,
                                          void* __restrict__ Outv,
                                          bf16_t* As, bf16_t* Bs, bool do_rope) {
    const int m0   = blockIdx.y * BM;
    const int n0   = blockIdx.x * BN;
    const int tid  = threadIdx.x;
    const int lane = tid & 31;
    const int wave = tid >> 5;
    const int mBase = (wave & 1) * 64;
    const int wn    = wave >> 1;
    const int nA    = (wn & 1) * 16 + (wn >> 1) * 64;   // first tile; partner at +32

    v8f acc[4][2];
#pragma unroll
    for (int i = 0; i < 4; ++i) { acc[i][0] = zero8(); acc[i][1] = zero8(); }

    const int srow = tid >> 1;          // 128 rows, 2 threads per row
    const int scol = (tid & 1) * 16;    // 16 bf16 elements (32B) per thread

    const uint32_t asBase = lds_off(As) + (uint32_t)(srow * LSTR + scol) * 2u;
    const uint32_t bsBase = lds_off(Bs) + (uint32_t)(srow * LSTR + scol) * 2u;
    const bf16_t* gA = Xb + (size_t)(m0 + srow) * DIM + scol;
    const bf16_t* gB = Wb + (size_t)(n0 + srow) * DIM + scol;

    // async-stage one (BMxBK A, BNxBK B) tile pair into buffer `buf`
    auto stage = [&](int k0, int buf) {
        const bf16_t* ga = gA + k0;
        const bf16_t* gb = gB + k0;
        const uint32_t la = asBase + (uint32_t)buf * TILE_BYTES;
        const uint32_t lb = bsBase + (uint32_t)buf * TILE_BYTES;
        async_copy16(la,      ga);
        async_copy16(la + 32, ga + 16);
        async_copy16(lb,      gb);
        async_copy16(lb + 32, gb + 16);
    };

    stage(0, 0);
    int idx = 0;
    for (int k0 = 0; k0 < DIM; k0 += BK, idx ^= 1) {
        wait_async0();        // my async copies for tile k0 have landed in LDS
        __syncthreads();      // everyone's have; everyone done reading buf idx^1
        if (k0 + BK < DIM) stage(k0 + BK, idx ^ 1);

        const bf16_t* curA = As + idx * (BM * LSTR);
        const bf16_t* curB = Bs + idx * (BM * LSTR);
        v16bf af[4];
#pragma unroll
        for (int mt = 0; mt < 4; ++mt)
            af[mt] = frag_a_lds(curA + (mBase + mt * 16) * LSTR, lane);
        v16bf bf0 = frag_b_lds(curB + nA * LSTR, lane);
        v16bf bf1 = frag_b_lds(curB + (nA + 32) * LSTR, lane);
#pragma unroll
        for (int mt = 0; mt < 4; ++mt) {
            acc[mt][0] = __builtin_amdgcn_wmma_f32_16x16x32_bf16(
                false, af[mt], false, bf0, (short)0, acc[mt][0], false, false);
            acc[mt][1] = __builtin_amdgcn_wmma_f32_16x16x32_bf16(
                false, af[mt], false, bf1, (short)0, acc[mt][1], false, false);
        }
    }

    // ---- epilogue: bias (+ optional RoPE), store ----
    const int rofs = (lane & 16) ? 8 : 0;
    const int cl   = lane & 15;
    const int c0   = n0 + nA + cl;
    const int c1   = c0 + 32;
    const float bi0 = bias[c0];
    const float bi1 = bias[c1];
#pragma unroll
    for (int mt = 0; mt < 4; ++mt) {
        const int rbase = m0 + mBase + mt * 16 + rofs;
        if (do_rope) {
            const int hd0 = c0 & 63;                        // in [0,32)
            const float theta = __expf((float)hd0 * -0.28782313662425576f); // 10000^(-hd0/32)
#pragma unroll
            for (int i = 0; i < 8; ++i) {
                const int row = rbase + i;
                const float pos = (float)(row & (SEQ - 1));
                float sn, cs;
                __sincosf(pos * theta, &sn, &cs);
                const float a0 = acc[mt][0][i] + bi0;
                const float a1 = acc[mt][1][i] + bi1;
                bf16_t* O = (bf16_t*)Outv;
                O[(size_t)row * DIM + c0] = (bf16_t)(a0 * cs - a1 * sn);
                O[(size_t)row * DIM + c1] = (bf16_t)(a1 * cs + a0 * sn);
            }
        } else {
#pragma unroll
            for (int i = 0; i < 8; ++i) {
                const int row = rbase + i;
                const float v0 = acc[mt][0][i] + bi0;
                const float v1 = acc[mt][1][i] + bi1;
                if constexpr (OUT_BF16) {
                    bf16_t* O = (bf16_t*)Outv;
                    O[(size_t)row * DIM + c0] = (bf16_t)v0;
                    O[(size_t)row * DIM + c1] = (bf16_t)v1;
                } else {
                    float* O = (float*)Outv;
                    O[(size_t)row * DIM + c0] = v0;
                    O[(size_t)row * DIM + c1] = v1;
                }
            }
        }
    }
}

__global__ __launch_bounds__(256) void qkv_rope_gemm(
    const bf16_t* __restrict__ X,
    const bf16_t* __restrict__ Wq, const bf16_t* __restrict__ Wk, const bf16_t* __restrict__ Wv,
    const float* __restrict__ bq, const float* __restrict__ bk, const float* __restrict__ bv,
    bf16_t* __restrict__ Oq, bf16_t* __restrict__ Ok, bf16_t* __restrict__ Ov) {
    __shared__ bf16_t As[2 * BM * LSTR];
    __shared__ bf16_t Bs[2 * BN * LSTR];
    const int z = blockIdx.z;
    const bf16_t* W    = (z == 0) ? Wq : ((z == 1) ? Wk : Wv);
    const float*  bias = (z == 0) ? bq : ((z == 1) ? bk : bv);
    bf16_t*       O    = (z == 0) ? Oq : ((z == 1) ? Ok : Ov);
    gemm_core<true>(X, W, bias, O, As, Bs, z != 2);
}

__global__ __launch_bounds__(256) void out_gemm(
    const bf16_t* __restrict__ A, const bf16_t* __restrict__ Wo,
    const float* __restrict__ bo, float* __restrict__ Out) {
    __shared__ bf16_t As[2 * BM * LSTR];
    __shared__ bf16_t Bs[2 * BN * LSTR];
    gemm_core<false>(A, Wo, bo, Out, As, Bs, false);
}

// ---- attention over the heads axis: one wave per (b,l) ---------------------
#define VSTR 72
#define PSTR 33
__global__ __launch_bounds__(128) void attn_kernel(
    const bf16_t* __restrict__ Q, const bf16_t* __restrict__ K,
    const bf16_t* __restrict__ V, bf16_t* __restrict__ O) {
    __shared__ bf16_t vt[4][32 * VSTR];
    __shared__ float  pt[4][32 * PSTR];
    const int wave = threadIdx.x >> 5;
    const int lane = threadIdx.x & 31;
    const int p = blockIdx.x * 4 + wave;
    const int b = p >> 12;          // / SEQ
    const int l = p & (SEQ - 1);
    const bf16_t* q = Q + (size_t)p * DIM;
    const bf16_t* k = K + (size_t)p * DIM;
    const bf16_t* v = V + (size_t)p * DIM;

    // stage V (32 heads x 64) into this wave's LDS region via async copies
    {
        const bf16_t* g = v + (size_t)lane * HDIM;
        const uint32_t d = lds_off(&vt[wave][lane * VSTR]);
#pragma unroll
        for (int j = 0; j < 8; ++j) async_copy16(d + j * 16, g + j * 8);
    }

    // scores = q . k^T  over Hd=64  -> 32x32, tiles 2x2, K tiles 2
    v8f sc[2][2];
    sc[0][0] = zero8(); sc[0][1] = zero8(); sc[1][0] = zero8(); sc[1][1] = zero8();
#pragma unroll
    for (int kt = 0; kt < 2; ++kt) {
        v16bf a0 = frag_a_gmem(q, 0,  kt, lane);
        v16bf a1 = frag_a_gmem(q, 16, kt, lane);
        v16bf b0 = frag_b_gmem(k, 0,  kt, lane);
        v16bf b1 = frag_b_gmem(k, 16, kt, lane);
        sc[0][0] = __builtin_amdgcn_wmma_f32_16x16x32_bf16(false, a0, false, b0, (short)0, sc[0][0], false, false);
        sc[0][1] = __builtin_amdgcn_wmma_f32_16x16x32_bf16(false, a0, false, b1, (short)0, sc[0][1], false, false);
        sc[1][0] = __builtin_amdgcn_wmma_f32_16x16x32_bf16(false, a1, false, b0, (short)0, sc[1][0], false, false);
        sc[1][1] = __builtin_amdgcn_wmma_f32_16x16x32_bf16(false, a1, false, b1, (short)0, sc[1][1], false, false);
    }

    // causal mask over head index + row softmax (16-lane shfl reductions)
    const int rofs = (lane & 16) ? 8 : 0;
    const int cl   = lane & 15;
#pragma unroll
    for (int mt = 0; mt < 2; ++mt) {
#pragma unroll
        for (int i = 0; i < 8; ++i) {
            const int row = mt * 16 + i + rofs;
            float s0 = sc[mt][0][i] * 0.125f;
            float s1 = sc[mt][1][i] * 0.125f;
            if (cl > row)      s0 = -1e9f;
            if (16 + cl > row) s1 = -1e9f;
            float mx = fmaxf(s0, s1);
#pragma unroll
            for (int d = 1; d < 16; d <<= 1) mx = fmaxf(mx, __shfl_xor(mx, d, 32));
            const float e0 = __expf(s0 - mx);
            const float e1 = __expf(s1 - mx);
            float sm = e0 + e1;
#pragma unroll
            for (int d = 1; d < 16; d <<= 1) sm += __shfl_xor(sm, d, 32);
            const float inv = 1.0f / sm;
            pt[wave][row * PSTR + cl]      = e0 * inv;
            pt[wave][row * PSTR + 16 + cl] = e1 * inv;
        }
    }

    // attn @ V : A = probs (32x32, K=heads), B = V (32x64)
    v16bf ap[2];
#pragma unroll
    for (int mt = 0; mt < 2; ++mt) {
        const float* pr = &pt[wave][(mt * 16 + cl) * PSTR];
        const int kk = (lane & 16) ? 16 : 0;
        v16bf a;
#pragma unroll
        for (int j = 0; j < 8; ++j) {
            a[j]     = (bf16_t)pr[kk + j];
            a[8 + j] = (bf16_t)pr[8 + kk + j];
        }
        ap[mt] = a;
    }

    wait_async0();   // V tile resident (this wave issued all copies for its region)

    v8f o2[2][4];
#pragma unroll
    for (int mt = 0; mt < 2; ++mt)
        for (int nt = 0; nt < 4; ++nt) o2[mt][nt] = zero8();
#pragma unroll
    for (int nt = 0; nt < 4; ++nt) {
        const int g0   = (lane & 16) ? 16 : 0;
        const int dcol = nt * 16 + cl;
        v16bf bvf;
#pragma unroll
        for (int j = 0; j < 16; ++j) bvf[j] = vt[wave][(g0 + j) * VSTR + dcol];
        o2[0][nt] = __builtin_amdgcn_wmma_f32_16x16x32_bf16(false, ap[0], false, bvf, (short)0, o2[0][nt], false, false);
        o2[1][nt] = __builtin_amdgcn_wmma_f32_16x16x32_bf16(false, ap[1], false, bvf, (short)0, o2[1][nt], false, false);
    }

    // store in the reference's transpose(0,2,1,3).reshape(B,L,D) layout:
    // flat = ((b*NH + h)*SEQ + l)*HDIM + hd
    const size_t LHD = (size_t)SEQ * HDIM;
#pragma unroll
    for (int mt = 0; mt < 2; ++mt)
#pragma unroll
        for (int nt = 0; nt < 4; ++nt)
#pragma unroll
            for (int i = 0; i < 8; ++i) {
                const int h  = mt * 16 + i + rofs;
                const int hd = nt * 16 + cl;
                const size_t f = ((size_t)b * NH + h) * LHD + (size_t)l * HDIM + hd;
                O[f] = (bf16_t)o2[mt][nt][i];
            }
}

extern "C" void kernel_launch(void* const* d_in, const int* in_sizes, int n_in,
                              void* d_out, int out_size, void* d_ws, size_t ws_size,
                              hipStream_t stream) {
    (void)in_sizes; (void)n_in; (void)out_size; (void)ws_size;
    const float* x  = (const float*)d_in[0];
    const float* Wq = (const float*)d_in[1];
    const float* bq = (const float*)d_in[2];
    const float* Wk = (const float*)d_in[3];
    const float* bk = (const float*)d_in[4];
    const float* Wv = (const float*)d_in[5];
    const float* bv = (const float*)d_in[6];
    const float* Wo = (const float*)d_in[7];
    const float* bo = (const float*)d_in[8];

    // workspace (bf16): XB 64Mi, 4 weights 8Mi each, Q/K/V/attn-out 64Mi each
    bf16_t* XB  = (bf16_t*)d_ws;
    bf16_t* WQB = XB  + (size_t)MTOT * DIM;
    bf16_t* WKB = WQB + (size_t)DIM * DIM;
    bf16_t* WVB = WKB + (size_t)DIM * DIM;
    bf16_t* WOB = WVB + (size_t)DIM * DIM;
    bf16_t* QB  = WOB + (size_t)DIM * DIM;
    bf16_t* KB  = QB  + (size_t)MTOT * DIM;
    bf16_t* VB  = KB  + (size_t)MTOT * DIM;
    bf16_t* OB  = VB  + (size_t)MTOT * DIM;

    const int nx8 = MTOT * DIM / 8;      // 4,194,304
    const int nw8 = DIM * DIM / 8;       // 524,288
    cvt_f32_bf16<<<nx8 / 256, 256, 0, stream>>>(x,  XB,  nx8);
    cvt_f32_bf16<<<nw8 / 256, 256, 0, stream>>>(Wq, WQB, nw8);
    cvt_f32_bf16<<<nw8 / 256, 256, 0, stream>>>(Wk, WKB, nw8);
    cvt_f32_bf16<<<nw8 / 256, 256, 0, stream>>>(Wv, WVB, nw8);
    cvt_f32_bf16<<<nw8 / 256, 256, 0, stream>>>(Wo, WOB, nw8);

    dim3 gq(DIM / BN, MTOT / BM, 3);
    qkv_rope_gemm<<<gq, 256, 0, stream>>>(XB, WQB, WKB, WVB, bq, bk, bv, QB, KB, VB);

    attn_kernel<<<MTOT / 4, 128, 0, stream>>>(QB, KB, VB, OB);

    dim3 go(DIM / BN, MTOT / BM, 1);
    out_gemm<<<go, 256, 0, stream>>>(OB, WOB, bo, (float*)d_out);
}
